// GNN_80590766342785
// MI455X (gfx1250) — compile-verified, matrix operand order checked by reading, not run on hardware
//
#include <hip/hip_runtime.h>

#define NNODES 50000
#define NEDGES 800000
#define FIN    1024
#define DH     128
#define BN_EPS 1e-5f

typedef __attribute__((ext_vector_type(16))) __bf16 bf16x16;
typedef __attribute__((ext_vector_type(8)))  float  v8f;

// K-offset map shared by A-fragment (row side) and B-fragment (col side),
// per CDNA5 ISA 7.12.2 16-bit 16x32 layout:
//   lanes 0-15 : VGPR0-3 -> K 0..7,  VGPR4-7 -> K 16..23
//   lanes 16-31: VGPR0-3 -> K 8..15, VGPR4-7 -> K 24..31
__device__ __forceinline__ int kmap(int lane_lo, int v) {
  return ((v < 4) ? (lane_lo ? 0 : 8) : (lane_lo ? 16 : 24)) + 2 * (v & 3);
}

// ---------------------------------------------------------------------------
// Pack a row-major fp32 weight [din x dout] into bf16, fragment-major:
// frag (kt, j) = 32 lanes x 16 bf16 contiguous (one WMMA B operand per frag).
// ---------------------------------------------------------------------------
__global__ void pack_w_kernel(const float* __restrict__ W, __bf16* __restrict__ out,
                              int din, int dout) {
  int tid = blockIdx.x * blockDim.x + threadIdx.x;
  int total = din * dout;
  if (tid >= total) return;
  int e    = tid & 15;
  int rem  = tid >> 4;
  int lane = rem & 31;
  rem >>= 5;
  int ntiles = dout >> 4;          // n-tiles of 16
  int j  = rem % ntiles;
  int kt = rem / ntiles;
  int lo = (lane < 16) ? 1 : 0;
  int v  = e >> 1, p = e & 1;
  int k  = kt * 32 + kmap(lo, v) + p;
  int n  = j * 16 + (lane & 15);
  out[tid] = (__bf16)W[k * dout + n];
}

// Load + convert one 16x32 fp32 A tile into a bf16 WMMA A fragment.
__device__ __forceinline__ bf16x16 load_a_frag(const float* __restrict__ ap, int lo) {
  bf16x16 a;
#pragma unroll
  for (int v = 0; v < 8; ++v) {
    float2 f = *(const float2*)(ap + kmap(lo, v));
    a[2 * v]     = (__bf16)f.x;
    a[2 * v + 1] = (__bf16)f.y;
  }
  return a;
}

// ---------------------------------------------------------------------------
// Fused dual GEMM: t = h @ W (raw), r = relu(h @ Wr + br).
// Block = 256 threads (8 waves), covers 128 rows. Waves 0-3 -> W on 32-row
// subtiles 0-3, waves 4-7 -> Wr on the same rows. Each wave owns a 32x128
// output tile (16 v8f accumulators); both 16-row halves share every B
// fragment, halving B-side L2 traffic vs a 16-row tile.
// __launch_bounds__(256, 1): release the full VGPR budget (1 wave/EU floor)
// so the 128 accumulator VGPRs + pipelined loads do NOT spill to scratch.
// ---------------------------------------------------------------------------
__global__ __launch_bounds__(256, 1)
void gemm_dual_kernel(const float* __restrict__ h, int K,
                      const __bf16* __restrict__ Wbf,
                      const __bf16* __restrict__ Wrbf,
                      const float* __restrict__ br,
                      float* __restrict__ t_out,
                      float* __restrict__ r_out) {
  const int lane  = threadIdx.x & 31;
  const int wave  = threadIdx.x >> 5;
  const int sub   = wave & 3;
  const int which = wave >> 2;           // 0 -> W/t, 1 -> Wr/r
  const int m0    = blockIdx.x * 128 + sub * 32;
  const int lo    = (lane < 16) ? 1 : 0;
  const int row0  = min(m0 +      (lane & 15), NNODES - 1);  // clamp; stores guarded
  const int row1  = min(m0 + 16 + (lane & 15), NNODES - 1);
  const float* ap0 = h + (long)row0 * K;
  const float* ap1 = h + (long)row1 * K;
  const __bf16* wsel = which ? Wrbf : Wbf;

  const v8f z = {0.f, 0.f, 0.f, 0.f, 0.f, 0.f, 0.f, 0.f};
  v8f acc[2][8];
#pragma unroll
  for (int m = 0; m < 2; ++m)
#pragma unroll
    for (int j = 0; j < 8; ++j) acc[m][j] = z;

  const int kts = K >> 5;
  for (int kt = 0; kt < kts; ++kt) {
    bf16x16 a0 = load_a_frag(ap0 + kt * 32, lo);
    bf16x16 a1 = load_a_frag(ap1 + kt * 32, lo);
    const __bf16* bp = wsel + ((long)(kt * 8) * 32 + lane) * 16;
#pragma unroll
    for (int j = 0; j < 8; ++j) {
      bf16x16 b = *(const bf16x16*)(bp + (long)j * 32 * 16);
      acc[0][j] = __builtin_amdgcn_wmma_f32_16x16x32_bf16(
          false, a0, false, b, (short)0, acc[0][j], false, false);
      acc[1][j] = __builtin_amdgcn_wmma_f32_16x16x32_bf16(
          false, a1, false, b, (short)0, acc[1][j], false, false);
    }
  }

  // Epilogue. C/D layout: lane<16 -> N=lane, M=base+v ; lane>=16 -> N=lane-16, M=base+8+v
  const int col = lane & 15;
#pragma unroll
  for (int m = 0; m < 2; ++m) {
    const int rbase = m0 + m * 16 + ((lane >> 4) << 3);
    if (which == 0) {
#pragma unroll
      for (int j = 0; j < 8; ++j) {
        int n = j * 16 + col;
#pragma unroll
        for (int v = 0; v < 8; ++v) {
          int r = rbase + v;
          if (r < NNODES) t_out[(long)r * DH + n] = acc[m][j][v];
        }
      }
    } else {
#pragma unroll
      for (int j = 0; j < 8; ++j) {
        int n = j * 16 + col;
        float bias = br[n];
#pragma unroll
        for (int v = 0; v < 8; ++v) {
          int r = rbase + v;
          if (r < NNODES) {
            float x = acc[m][j][v] + bias;
            r_out[(long)r * DH + n] = x > 0.f ? x : 0.f;
          }
        }
      }
    }
  }
}

// ---------------------------------------------------------------------------
__global__ void zero_kernel(float* __restrict__ p, long n) {
  long i = (long)blockIdx.x * blockDim.x + threadIdx.x;
  long stride = (long)gridDim.x * blockDim.x;
  for (; i < n; i += stride) p[i] = 0.f;
}

// One wave per edge: agg[dst] += t[src] over 128 feats (L2-resident atomics).
// Gather side vectorized as one b128 load per lane.
__global__ void scatter_kernel(const int* __restrict__ src, const int* __restrict__ dst,
                               const float* __restrict__ t, float* __restrict__ agg) {
  int edge = (int)((blockIdx.x * (long)blockDim.x + threadIdx.x) >> 5);
  int lane = threadIdx.x & 31;
  if (edge >= NEDGES) return;
  long s = (long)src[edge] * DH;
  long d = (long)dst[edge] * DH;
  float4 v = *(const float4*)(t + s + lane * 4);
  float* ap = agg + d + lane * 4;
  atomicAdd(ap + 0, v.x);
  atomicAdd(ap + 1, v.y);
  atomicAdd(ap + 2, v.z);
  atomicAdd(ap + 3, v.w);
}

// new = relu(agg + b) + r ; h = BN_eval(new)
__global__ void post_kernel(const float* __restrict__ agg, const float* __restrict__ r,
                            const float* __restrict__ b, const float* __restrict__ gamma,
                            const float* __restrict__ beta, const float* __restrict__ mean,
                            const float* __restrict__ var, float* __restrict__ hout) {
  long i = (long)blockIdx.x * blockDim.x + threadIdx.x;
  if (i >= (long)NNODES * DH) return;
  int c = (int)(i & (DH - 1));
  float x = agg[i] + b[c];
  x = x > 0.f ? x : 0.f;
  x += r[i];
  hout[i] = (x - mean[c]) * rsqrtf(var[c] + BN_EPS) * gamma[c] + beta[c];
}

// One wave per node: logits = h @ Wd + bd, softmax over 2 classes.
__global__ void head_kernel(const float* __restrict__ h, const float* __restrict__ Wd,
                            const float* __restrict__ bd, float* __restrict__ out) {
  int node = (int)((blockIdx.x * (long)blockDim.x + threadIdx.x) >> 5);
  int lane = threadIdx.x & 31;
  if (node >= NNODES) return;
  const float* hp = h + (long)node * DH;
  float s0 = 0.f, s1 = 0.f;
#pragma unroll
  for (int c = lane; c < DH; c += 32) {
    float hv = hp[c];
    s0 += hv * Wd[c * 2 + 0];
    s1 += hv * Wd[c * 2 + 1];
  }
#pragma unroll
  for (int off = 16; off >= 1; off >>= 1) {
    s0 += __shfl_xor(s0, off, 32);
    s1 += __shfl_xor(s1, off, 32);
  }
  if (lane == 0) {
    s0 += bd[0];
    s1 += bd[1];
    float m  = fmaxf(s0, s1);
    float e0 = __expf(s0 - m), e1 = __expf(s1 - m);
    float inv = 1.f / (e0 + e1);
    out[node * 2 + 0] = e0 * inv;
    out[node * 2 + 1] = e1 * inv;
  }
}

// ---------------------------------------------------------------------------
extern "C" void kernel_launch(void* const* d_in, const int* in_sizes, int n_in,
                              void* d_out, int out_size, void* d_ws, size_t ws_size,
                              hipStream_t stream) {
  const float* in_feat = (const float*)d_in[0];
  const int*   src     = (const int*)d_in[1];
  const int*   dst     = (const int*)d_in[2];
  const float* W0      = (const float*)d_in[3];
  const float* b0      = (const float*)d_in[4];
  const float* Wr0     = (const float*)d_in[5];
  const float* br0     = (const float*)d_in[6];
  const float* gamma0  = (const float*)d_in[7];
  const float* beta0   = (const float*)d_in[8];
  const float* mean0   = (const float*)d_in[9];
  const float* var0    = (const float*)d_in[10];
  const float* W1      = (const float*)d_in[11];
  const float* b1      = (const float*)d_in[12];
  const float* Wr1     = (const float*)d_in[13];
  const float* br1     = (const float*)d_in[14];
  const float* gamma1  = (const float*)d_in[15];
  const float* beta1   = (const float*)d_in[16];
  const float* mean1   = (const float*)d_in[17];
  const float* var1    = (const float*)d_in[18];
  const float* Wd      = (const float*)d_in[19];
  const float* bd      = (const float*)d_in[20];
  float* out = (float*)d_out;

  // Workspace layout
  char* ws = (char*)d_ws;
  __bf16* W0bf  = (__bf16*)(ws + 0);          // 1024*128*2 = 256 KB
  __bf16* Wr0bf = (__bf16*)(ws + 262144);     // 256 KB
  __bf16* W1bf  = (__bf16*)(ws + 524288);     //  32 KB
  __bf16* Wr1bf = (__bf16*)(ws + 557056);     //  32 KB
  const long NH = (long)NNODES * DH;          // 6.4M floats = 25.6 MB each
  float* bufA = (float*)(ws + (1 << 20));     // t0, then h1
  float* bufB = bufA + NH;                    // r0, then t1, then h2
  float* bufC = bufB + NH;                    // agg (both layers)
  float* bufD = bufC + NH;                    // r1

  const int TB = 256;
  // 1) pack weights -> bf16 WMMA-fragment layout
  pack_w_kernel<<<(FIN * DH + TB - 1) / TB, TB, 0, stream>>>(W0,  W0bf,  FIN, DH);
  pack_w_kernel<<<(FIN * DH + TB - 1) / TB, TB, 0, stream>>>(Wr0, Wr0bf, FIN, DH);
  pack_w_kernel<<<(DH * DH + TB - 1) / TB,  TB, 0, stream>>>(W1,  W1bf,  DH,  DH);
  pack_w_kernel<<<(DH * DH + TB - 1) / TB,  TB, 0, stream>>>(Wr1, Wr1bf, DH,  DH);

  const int gemm_blocks = (NNODES + 127) / 128;   // 128 rows per block
  const int scat_blocks = (NEDGES + 7) / 8;       // 8 edges (waves) per block
  const int post_blocks = (int)((NH + TB - 1) / TB);
  const int head_blocks = (NNODES + 7) / 8;

  // ---- layer 0 ----
  zero_kernel<<<1024, TB, 0, stream>>>(bufC, NH);
  gemm_dual_kernel<<<gemm_blocks, TB, 0, stream>>>(in_feat, FIN, W0bf, Wr0bf, br0,
                                                   bufA /*t0*/, bufB /*r0*/);
  scatter_kernel<<<scat_blocks, TB, 0, stream>>>(src, dst, bufA, bufC);
  post_kernel<<<post_blocks, TB, 0, stream>>>(bufC, bufB, b0, gamma0, beta0, mean0, var0,
                                              bufA /*h1 (t0 dead)*/);

  // ---- layer 1 ----
  zero_kernel<<<1024, TB, 0, stream>>>(bufC, NH);
  gemm_dual_kernel<<<gemm_blocks, TB, 0, stream>>>(bufA, DH, W1bf, Wr1bf, br1,
                                                   bufB /*t1 (r0 dead)*/, bufD /*r1*/);
  scatter_kernel<<<scat_blocks, TB, 0, stream>>>(src, dst, bufB, bufC);
  post_kernel<<<post_blocks, TB, 0, stream>>>(bufC, bufD, b1, gamma1, beta1, mean1, var1,
                                              bufB /*h2 (t1 dead)*/);

  // ---- head ----
  head_kernel<<<head_blocks, TB, 0, stream>>>(bufB, Wd, bd, out);
}
